// TSK_46222438039894
// MI455X (gfx1250) — compile-verified, matrix operand order checked by reading, not run on hardware
//
#include <hip/hip_runtime.h>

typedef __attribute__((ext_vector_type(16))) _Float16     v16h;
typedef __attribute__((ext_vector_type(8)))  float        v8f;
typedef __attribute__((ext_vector_type(4)))  float        v4f;
typedef __attribute__((ext_vector_type(2)))  float        v2f;
typedef __attribute__((ext_vector_type(4)))  unsigned int v4u;
typedef __attribute__((ext_vector_type(2)))  unsigned int v2u;

#define N_BATCH 8192
#define D_IN    256
#define N_RULE  32
#define N_OUT   32

// ---------------- ws layout ----------------
// abfrag : f32, 256 frags x 64 floats        = 65536 B   (logit B-operands: a=1/(2s^2), b=-c/s^2)
// cc     : f32, 32                           = 128 B (padded to 256)
// pfrag  : f16, 512 frags x 512 halves       = 524288 B  (main GEMM B-operands, con_param[:, :, 1:])
// bfrag  : f16, 2 frags x 512 halves         = 2048 B    (bias GEMM B-operand, con_param[:, :, 0])
#define WS_OFF_AB   0
#define WS_OFF_CC   65536
#define WS_OFF_P    (65536 + 256)
#define WS_OFF_B    (WS_OFF_P + 524288)

// k-position owned by (lane, vgpr v, half h) in a 16-bit 16x32 A / 32x16 B fragment
__device__ __host__ inline int k16(int lane, int v, int h) {
    return ((v < 4) ? 2 * v : 16 + 2 * (v - 4)) + ((lane >= 16) ? 8 : 0) + h;
}

// ---- prep: f32 B-fragments for the logit GEMMs (16x16x4 layout: k = (lane>=16?2:0)+v) ----
__global__ void prep_ab(const double* __restrict__ center, const double* __restrict__ spread,
                        float* __restrict__ abfrag) {
    int t = blockIdx.x * blockDim.x + threadIdx.x;   // 0..16383
    if (t >= 256 * 64) return;
    int fa = t >> 6, e = t & 63;
    int lane = e >> 1, v = e & 1;
    int which = fa & 1, ot = (fa >> 1) & 1, dblk = fa >> 2;
    int k = ((lane >= 16) ? 2 : 0) + v;
    int d = dblk * 4 + k;
    int r = (lane & 15) + ot * 16;
    double s = spread[r * D_IN + d];
    double c = center[r * D_IN + d];
    double val = (which == 0) ? 1.0 / (2.0 * s * s) : -c / (s * s);
    abfrag[t] = (float)val;
}

__global__ void prep_cc(const double* __restrict__ center, const double* __restrict__ spread,
                        float* __restrict__ cc) {
    int r = threadIdx.x;
    if (r >= N_RULE) return;
    double acc = 0.0;
    for (int d = 0; d < D_IN; ++d) {
        double s = spread[r * D_IN + d], c = center[r * D_IN + d];
        acc += c * c / (2.0 * s * s);
    }
    cc[r] = (float)acc;
}

// ---- prep: f16 B-fragments for the main GEMM ----
__global__ void prep_p(const double* __restrict__ con, _Float16* __restrict__ pfrag) {
    int t = blockIdx.x * blockDim.x + threadIdx.x;   // 0..262143
    if (t >= 512 * 512) return;
    int id = t >> 9, e = t & 511;
    int lane = e >> 4, q = e & 15, v = q >> 1, h = q & 1;
    int ot = id & 1, dblk = (id >> 1) & 7, r = id >> 4;
    int k = k16(lane, v, h);
    int d = dblk * 32 + k;
    int o = (lane & 15) + ot * 16;
    pfrag[t] = (_Float16)con[((size_t)o * N_RULE + r) * (D_IN + 1) + 1 + d];
}

// ---- prep: f16 B-fragment for the bias GEMM (k = rule index) ----
__global__ void prep_bias(const double* __restrict__ con, _Float16* __restrict__ bfrag) {
    int t = blockIdx.x * blockDim.x + threadIdx.x;   // 0..1023
    if (t >= 2 * 512) return;
    int id = t >> 9, e = t & 511;
    int lane = e >> 4, q = e & 15, v = q >> 1, h = q & 1;
    int k = k16(lane, v, h);                         // rule
    int o = (lane & 15) + id * 16;
    bfrag[t] = (_Float16)con[((size_t)o * N_RULE + k) * (D_IN + 1)];
}

// ---------------- main fused kernel ----------------
// 64 threads = 2 waves; each wave owns a 16-row batch tile.
__global__ void __launch_bounds__(64) tsk_main(
    const float* __restrict__ x,
    const float* __restrict__ abfrag,
    const float* __restrict__ cc,
    const _Float16* __restrict__ pfrag,
    const _Float16* __restrict__ bfrag,
    float* __restrict__ out)
{
    __shared__ __align__(16) float    xf[2][16][D_IN];
    __shared__ __align__(16) _Float16 xh[2][16][D_IN];
    __shared__ __align__(16) float    wl[2][16][N_RULE];

    const int lane    = threadIdx.x & 31;
    const int wave    = threadIdx.x >> 5;
    const int rowbase = blockIdx.x * 32 + wave * 16;
    const int halfsel = (lane >= 16) ? 1 : 0;
    const int col     = lane & 15;

    // ---- stage 0: stage x tile into LDS (f32 + f16 copies) ----
    for (int i = 0; i < 32; ++i) {
        int idx = i * 32 + lane;          // float4 index in 16x256 tile
        int row = idx >> 6;
        int c4  = idx & 63;
        v4f g = *(const v4f*)(x + (size_t)(rowbase + row) * D_IN + c4 * 4);
        *(v4f*)&xf[wave][row][c4 * 4] = g;
        union { unsigned int u[2]; _Float16 hh[4]; } hp;
        hp.hh[0] = (_Float16)g[0]; hp.hh[1] = (_Float16)g[1];
        hp.hh[2] = (_Float16)g[2]; hp.hh[3] = (_Float16)g[3];
        *(v2u*)&xh[wave][row][c4 * 4] = (v2u){hp.u[0], hp.u[1]};
    }
    __syncthreads();

    // ---- stage 1: logits via f32 WMMA: l = -(x^2 @ a + x @ b + cc) ----
    v8f l0 = {}; v8f l1 = {};
    for (int db = 0; db < 64; ++db) {
        v2f xa  = *(const v2f*)&xf[wave][col][db * 4 + halfsel * 2];
        v2f x2a = xa * xa;
        const float* ab = abfrag;
        v2f bA0 = *(const v2f*)(ab + ((db * 2 + 0) * 2 + 0) * 64 + lane * 2);
        v2f bB0 = *(const v2f*)(ab + ((db * 2 + 0) * 2 + 1) * 64 + lane * 2);
        v2f bA1 = *(const v2f*)(ab + ((db * 2 + 1) * 2 + 0) * 64 + lane * 2);
        v2f bB1 = *(const v2f*)(ab + ((db * 2 + 1) * 2 + 1) * 64 + lane * 2);
        l0 = __builtin_amdgcn_wmma_f32_16x16x4_f32(false, x2a, false, bA0, (short)0, l0, false, false);
        l0 = __builtin_amdgcn_wmma_f32_16x16x4_f32(false, xa,  false, bB0, (short)0, l0, false, false);
        l1 = __builtin_amdgcn_wmma_f32_16x16x4_f32(false, x2a, false, bA1, (short)0, l1, false, false);
        l1 = __builtin_amdgcn_wmma_f32_16x16x4_f32(false, xa,  false, bB1, (short)0, l1, false, false);
    }

    // ---- softmax over 32 rules (rows in C layout: m = v + 8*halfsel) ----
    {
        float cc0 = cc[col], cc1 = cc[col + 16];
        float lg0[8], lg1[8], m[8], e0[8], e1[8], s[8];
        for (int v = 0; v < 8; ++v) { lg0[v] = -(l0[v] + cc0); lg1[v] = -(l1[v] + cc1); }
        for (int v = 0; v < 8; ++v) m[v] = fmaxf(lg0[v], lg1[v]);
        for (int msk = 1; msk <= 8; msk <<= 1)
            for (int v = 0; v < 8; ++v) m[v] = fmaxf(m[v], __shfl_xor(m[v], msk, 32));
        for (int v = 0; v < 8; ++v) {
            e0[v] = expf(lg0[v] - m[v]);
            e1[v] = expf(lg1[v] - m[v]);
            s[v]  = e0[v] + e1[v];
        }
        for (int msk = 1; msk <= 8; msk <<= 1)
            for (int v = 0; v < 8; ++v) s[v] += __shfl_xor(s[v], msk, 32);
        for (int v = 0; v < 8; ++v) {
            float inv = 1.0f / s[v];
            wl[wave][v + 8 * halfsel][col]      = e0[v] * inv;
            wl[wave][v + 8 * halfsel][col + 16] = e1[v] * inv;
        }
    }
    __syncthreads();

    // ---- stage 2: build f16 A-fragments (row = lane&15, k-runs of 8) ----
    union Frag { v16h h; v4u u4[2]; };
    Frag af[8];
    const int kb = halfsel * 8;
    for (int db = 0; db < 8; ++db) {
        af[db].u4[0] = *(const v4u*)&xh[wave][col][db * 32 + kb];
        af[db].u4[1] = *(const v4u*)&xh[wave][col][db * 32 + kb + 16];
    }

    // ---- stage 3: per-rule GEMM (f16 WMMA) + f32 w-weighted accumulation ----
    v8f acc0 = {}; v8f acc1 = {};
    for (int r = 0; r < N_RULE; ++r) {
        if (r + 1 < N_RULE)
            __builtin_prefetch(pfrag + (size_t)(((r + 1) * 8) * 2) * 512 + lane * 16, 0, 0);
        v8f d0 = {}; v8f d1 = {};
        for (int db = 0; db < 8; ++db) {
            const _Float16* p0 = pfrag + (size_t)(((r * 8 + db) * 2 + 0)) * 512 + lane * 16;
            const _Float16* p1 = pfrag + (size_t)(((r * 8 + db) * 2 + 1)) * 512 + lane * 16;
            Frag b0, b1;
            b0.u4[0] = *(const v4u*)p0; b0.u4[1] = *(const v4u*)(p0 + 8);
            b1.u4[0] = *(const v4u*)p1; b1.u4[1] = *(const v4u*)(p1 + 8);
            d0 = __builtin_amdgcn_wmma_f32_16x16x32_f16(false, af[db].h, false, b0.h, (short)0, d0, false, false);
            d1 = __builtin_amdgcn_wmma_f32_16x16x32_f16(false, af[db].h, false, b1.h, (short)0, d1, false, false);
        }
        for (int v = 0; v < 8; ++v) {
            float wv = wl[wave][v + 8 * halfsel][r];   // LDS broadcast per half-wave
            acc0[v] = fmaf(wv, d0[v], acc0[v]);
            acc1[v] = fmaf(wv, d1[v], acc1[v]);
        }
    }

    // ---- stage 4: bias GEMM  out += w[16x32] @ b[32x32]  (2 WMMAs) ----
    {
        v16h wa;
        for (int j = 0; j < 8; ++j) {
            wa[j]     = (_Float16)wl[wave][col][kb + j];
            wa[8 + j] = (_Float16)wl[wave][col][kb + 16 + j];
        }
        const _Float16* q0 = bfrag + lane * 16;
        const _Float16* q1 = bfrag + 512 + lane * 16;
        Frag bb0, bb1;
        bb0.u4[0] = *(const v4u*)q0; bb0.u4[1] = *(const v4u*)(q0 + 8);
        bb1.u4[0] = *(const v4u*)q1; bb1.u4[1] = *(const v4u*)(q1 + 8);
        acc0 = __builtin_amdgcn_wmma_f32_16x16x32_f16(false, wa, false, bb0.h, (short)0, acc0, false, false);
        acc1 = __builtin_amdgcn_wmma_f32_16x16x32_f16(false, wa, false, bb1.h, (short)0, acc1, false, false);
    }

    // ---- store (C layout: vgpr v -> row v + 8*halfsel, col = lane&15 + 16*ot) ----
    for (int v = 0; v < 8; ++v) {
        int row = rowbase + v + 8 * halfsel;
        out[(size_t)row * N_OUT + col]      = acc0[v];
        out[(size_t)row * N_OUT + col + 16] = acc1[v];
    }
}

extern "C" void kernel_launch(void* const* d_in, const int* in_sizes, int n_in,
                              void* d_out, int out_size, void* d_ws, size_t ws_size,
                              hipStream_t stream) {
    const float*  x      = (const float*)d_in[0];
    const double* center = (const double*)d_in[1];
    const double* spread = (const double*)d_in[2];
    const double* con    = (const double*)d_in[3];
    float* out = (float*)d_out;

    char* ws = (char*)d_ws;
    float*    abfrag = (float*)(ws + WS_OFF_AB);
    float*    ccp    = (float*)(ws + WS_OFF_CC);
    _Float16* pfrag  = (_Float16*)(ws + WS_OFF_P);
    _Float16* bfrag  = (_Float16*)(ws + WS_OFF_B);

    prep_ab  <<<64,   256, 0, stream>>>(center, spread, abfrag);
    prep_cc  <<<1,    32,  0, stream>>>(center, spread, ccp);
    prep_p   <<<1024, 256, 0, stream>>>(con, pfrag);
    prep_bias<<<4,    256, 0, stream>>>(con, bfrag);

    tsk_main<<<N_BATCH / 32, 64, 0, stream>>>(x, abfrag, ccp, pfrag, bfrag, out);
}
